// Convpass_hypernet_mask_serial_22359599743399
// MI455X (gfx1250) — compile-verified, matrix operand order checked by reading, not run on hardware
//
#include <hip/hip_runtime.h>

// ---------- types ----------
typedef __attribute__((ext_vector_type(16))) __bf16 bf16x16;
typedef __attribute__((ext_vector_type(8)))  float  f32x8;

// ---------- helpers ----------
__device__ __forceinline__ __bf16 f2bf(float f) { return (__bf16)f; }
__device__ __forceinline__ float  bf2f(__bf16 h) { return (float)h; }
__device__ __forceinline__ float qgelu_f(float v) {
  return v / (1.0f + __expf(-1.702f * v));
}
__device__ __forceinline__ f32x8 wmma_bf16(bf16x16 a, bf16x16 b, f32x8 c) {
  return __builtin_amdgcn_wmma_f32_16x16x32_bf16(false, a, false, b, (short)0, c,
                                                 false, false);
}
// 16x32 bf16 fragment (A layout; B tiles are stored (N,K) so same loader works).
// lane: row m = tid&15; K-chunk select by lane bit4 (ISA 7.12.2 16-bit A table).
__device__ __forceinline__ bf16x16 frag_ld_bf16(const __bf16* rowp) {
  int ks = ((threadIdx.x >> 4) & 1) << 3;
  const __bf16* p = rowp + ks;
  bf16x16 r;
#pragma unroll
  for (int e = 0; e < 8; ++e) r[e] = p[e];
#pragma unroll
  for (int e = 0; e < 8; ++e) r[8 + e] = p[16 + e];
  return r;
}
__device__ __forceinline__ bf16x16 frag_bf16(const __bf16* base, int ld) {
  return frag_ld_bf16(base + (int)(threadIdx.x & 15) * ld);
}
__device__ __forceinline__ bf16x16 frag_f32(const float* base, int ld) {
  const float* p = base + (int)(threadIdx.x & 15) * ld + (((threadIdx.x >> 4) & 1) << 3);
  bf16x16 r;
#pragma unroll
  for (int e = 0; e < 8; ++e) r[e] = f2bf(p[e]);
#pragma unroll
  for (int e = 0; e < 8; ++e) r[8 + e] = f2bf(p[16 + e]);
  return r;
}

// ---------- K1: fused input projections ----------
// pf = relu(x@W1^T+b1)@W2^T+b2 ; xd = qgelu(x@down^T+db). One pass over x.
// LDS-staged bf16 tiles: x chunk (64x128) + fused weight chunk (128x128).
__global__ void __launch_bounds__(512) k_fused_in(
    const float* __restrict__ x, const float* __restrict__ mw1,
    const float* __restrict__ mb1, const float* __restrict__ mw2,
    const float* __restrict__ mb2, const float* __restrict__ dw,
    const float* __restrict__ db, float* __restrict__ pf,
    float* __restrict__ xd) {
  __shared__ __bf16 XT[64][128];    // 16 KB
  __shared__ __bf16 WT[128][128];   // 32 KB  rows 0-63: meta_w1, 64-127: down_w
  __shared__ __bf16 T1[64][64];     // 8 KB
  const int r0 = blockIdx.x * 64;
  const int tid = threadIdx.x;
  const int w = tid >> 5, lane = tid & 31;
  const int rt = w & 3, nh = w >> 2;                 // 4 row-tiles x 4 col-halves(32)
  const int cm = (lane >> 4) << 3, cn = lane & 15;

  f32x8 acc[2] = {};
  for (int ck = 0; ck < 6; ++ck) {                   // K = 768 in 6 chunks of 128
    for (int idx = tid; idx < 8192; idx += 512) {    // stage x -> bf16
      int r = idx >> 7, k = idx & 127;
      const float* src = x + (size_t)(r0 + r) * 768 + ck * 128 + k;
      XT[r][k] = f2bf(*src);
      if (ck < 5) __builtin_prefetch(src + 128, 0, 1);  // global_prefetch_b8
    }
    for (int idx = tid; idx < 16384; idx += 512) {   // stage [W1;down] -> bf16
      int n = idx >> 7, k = idx & 127;
      const float* src = (n < 64) ? (mw1 + (size_t)n * 768)
                                  : (dw + (size_t)(n - 64) * 768);
      WT[n][k] = f2bf(src[ck * 128 + k]);
    }
    __syncthreads();
#pragma unroll
    for (int kt = 0; kt < 4; ++kt) {
      bf16x16 af = frag_bf16(&XT[rt * 16][kt * 32], 128);
#pragma unroll
      for (int j = 0; j < 2; ++j) {
        bf16x16 bf = frag_bf16(&WT[nh * 32 + j * 16][kt * 32], 128);
        acc[j] = wmma_bf16(af, bf, acc[j]);
      }
    }
    __syncthreads();
  }
  if (nh < 2) {                                      // meta mid -> relu -> LDS (bf16)
#pragma unroll
    for (int j = 0; j < 2; ++j) {
      int col = nh * 32 + j * 16 + cn;
      float bias = mb1[col];
#pragma unroll
      for (int v = 0; v < 8; ++v) {
        int m = rt * 16 + cm + v;
        float t = acc[j][v] + bias;
        T1[m][col] = f2bf(t > 0.f ? t : 0.f);
      }
    }
  } else {                                           // down path -> qgelu -> xd
#pragma unroll
    for (int j = 0; j < 2; ++j) {
      int nd = (nh - 2) * 32 + j * 16 + cn;
      float bias = db[nd];
#pragma unroll
      for (int v = 0; v < 8; ++v) {
        int row = r0 + rt * 16 + cm + v;
        xd[(size_t)row * 64 + nd] = qgelu_f(acc[j][v] + bias);
      }
    }
  }
  __syncthreads();
  // second GEMM: pf = T1 @ mw2^T + mb2   (16 waves -> 16 tiles)
  {
    const int rt2 = w & 3, nt2 = w >> 2;
    f32x8 c2 = {};
#pragma unroll
    for (int kt = 0; kt < 2; ++kt) {
      bf16x16 af = frag_bf16(&T1[rt2 * 16][0] + kt * 32, 64);
      bf16x16 bf = frag_f32(mw2 + (size_t)(nt2 * 16) * 64 + kt * 32, 64);
      c2 = wmma_bf16(af, bf, c2);
    }
    int col = nt2 * 16 + cn;
    float bias = mb2[col];
#pragma unroll
    for (int v = 0; v < 8; ++v) {
      int row = r0 + rt2 * 16 + cm + v;
      pf[(size_t)row * 64 + col] = c2[v] + bias;
    }
  }
}

// ---------- K2: mask softmax + feat (tiny, one block per batch) ----------
__global__ void __launch_bounds__(256) k_mask_feat(
    const float* __restrict__ pf, const float* __restrict__ mtw_g,
    float* __restrict__ feat) {
  __shared__ float lg[4][576];
  __shared__ float sL[576];
  __shared__ float mtw[4][64];
  __shared__ float red[256];
  __shared__ float mxs[4], invs[4];
  const int b = blockIdx.x;
  const int tid = threadIdx.x;
  mtw[tid >> 6][tid & 63] = mtw_g[tid];
  __syncthreads();
  const float* tok = pf + ((size_t)b * 577 + 1) * 64;
  for (int l = tid; l < 576; l += 256) {
    const float* row = tok + (size_t)l * 64;
    float d0 = 0, d1 = 0, d2 = 0, d3 = 0;
    for (int h = 0; h < 64; ++h) {
      float xv = row[h];
      d0 += xv * mtw[0][h]; d1 += xv * mtw[1][h];
      d2 += xv * mtw[2][h]; d3 += xv * mtw[3][h];
    }
    lg[0][l] = d0; lg[1][l] = d1; lg[2][l] = d2; lg[3][l] = d3;
  }
  __syncthreads();
  for (int m = 0; m < 4; ++m) {
    float v = -3.4e38f;
    for (int l = tid; l < 576; l += 256) v = fmaxf(v, lg[m][l]);
    red[tid] = v; __syncthreads();
    for (int s = 128; s > 0; s >>= 1) {
      if (tid < s) red[tid] = fmaxf(red[tid], red[tid + s]);
      __syncthreads();
    }
    float m_max = red[0]; __syncthreads();
    float sv = 0;
    for (int l = tid; l < 576; l += 256) sv += __expf(lg[m][l] - m_max);
    red[tid] = sv; __syncthreads();
    for (int s = 128; s > 0; s >>= 1) {
      if (tid < s) red[tid] += red[tid + s];
      __syncthreads();
    }
    if (tid == 0) { mxs[m] = m_max; invs[m] = 1.0f / red[0]; }
    __syncthreads();
  }
  for (int l = tid; l < 576; l += 256) {
    float t = 0;
    for (int m = 0; m < 4; ++m) t += __expf(lg[m][l] - mxs[m]) * invs[m];
    sL[l] = t;
  }
  __syncthreads();
  const int h = tid & 63, g = tid >> 6;
  float a = 0;
  for (int l = g; l < 576; l += 4) a += sL[l] * tok[(size_t)l * 64 + h];
  red[tid] = a; __syncthreads();
  if (tid < 64)
    feat[(size_t)b * 64 + tid] =
        red[tid] + red[tid + 64] + red[tid + 128] + red[tid + 192];
}

// ---------- K3: hypernet GEMM  w = feat @ hyper_w^T + hb ----------
// stores bf16 in permuted layout wperm[b][tap(ki*3+kj)][dout][din]
__global__ void __launch_bounds__(512) k_hyper(
    const float* __restrict__ feat, const float* __restrict__ hw,
    const float* __restrict__ hb, __bf16* __restrict__ wperm) {
  __shared__ __bf16 FL[64][64];
  const int tid = threadIdx.x;
  for (int idx = tid; idx < 4096; idx += 512) (&FL[0][0])[idx] = f2bf(feat[idx]);
  __syncthreads();
  const int n0 = blockIdx.x * 128;
  const int w = tid >> 5, lane = tid & 31;
  const int rt = w & 3, np = w >> 2;
  const int cm = (lane >> 4) << 3, cn = lane & 15;
  f32x8 acc[2] = {};
#pragma unroll
  for (int kt = 0; kt < 2; ++kt) {
    bf16x16 af = frag_bf16(&FL[rt * 16][0] + kt * 32, 64);
#pragma unroll
    for (int j = 0; j < 2; ++j) {
      bf16x16 bf = frag_f32(hw + (size_t)(n0 + np * 32 + j * 16) * 64 + kt * 32, 64);
      acc[j] = wmma_bf16(af, bf, acc[j]);
    }
  }
#pragma unroll
  for (int j = 0; j < 2; ++j) {
    int ng = n0 + np * 32 + j * 16 + cn;     // n = ((dout*64+din)*3+ki)*3+kj
    int kj = ng % 3, t = ng / 3;
    int ki = t % 3;  t /= 3;
    int din = t & 63, dout = t >> 6;
    float bias = hb[ng];
#pragma unroll
    for (int v = 0; v < 8; ++v) {
      int bidx = rt * 16 + cm + v;           // batch
      wperm[(((size_t)bidx * 9 + ki * 3 + kj) * 64 + dout) * 64 + din] =
          f2bf(acc[j][v] + bias);
    }
  }
}

// ---------- K4: per-batch 3x3 conv (implicit GEMM) + CLS matvec, qgelu ----------
__global__ void __launch_bounds__(256) k_conv_cls(
    const float* __restrict__ xd, const __bf16* __restrict__ wperm,
    __bf16* __restrict__ y) {
  __shared__ __bf16 Xs[10][26][64];   // 8 output rows + halo, zero-padded cols
  __shared__ float red[256];
  const int b = blockIdx.x, my = blockIdx.y;
  const int tid = threadIdx.x;
  if (my < 3) {
    const int i0 = my * 8;
    for (int idx = tid; idx < 10 * 26 * 64; idx += 256) {
      int c = idx & 63, rest = idx >> 6;
      int col = rest % 26, row = rest / 26;
      int gi = i0 - 1 + row, gj = col - 1;
      float v = 0.0f;
      if (gi >= 0 && gi < 24 && gj >= 0 && gj < 24)
        v = xd[((size_t)b * 577 + 1 + gi * 24 + gj) * 64 + c];
      Xs[row][col][c] = f2bf(v);
    }
    __syncthreads();
    const int w = tid >> 5, lane = tid & 31;
    const int nt = w & 3, mg = w >> 2;          // 4 n-tiles x 2 m-groups(6 tiles)
    const int cm = (lane >> 4) << 3, cn = lane & 15;
    f32x8 acc[6] = {};
    const __bf16* wb = wperm + (size_t)b * 9 * 64 * 64;
    for (int tap = 0; tap < 9; ++tap) {
      int ki = tap / 3, kj = tap - ki * 3;
#pragma unroll
      for (int kt = 0; kt < 2; ++kt) {
        bf16x16 bf = frag_bf16(wb + ((size_t)tap * 64 + nt * 16) * 64 + kt * 32, 64);
#pragma unroll
        for (int t6 = 0; t6 < 6; ++t6) {
          int P = (mg * 6 + t6) * 16 + (lane & 15);   // local position 0..191
          int il = P / 24, jc = P - il * 24;
          bf16x16 af = frag_ld_bf16(&Xs[il + ki][jc + kj][0] + kt * 32);
          acc[t6] = wmma_bf16(af, bf, acc[t6]);
        }
      }
    }
#pragma unroll
    for (int t6 = 0; t6 < 6; ++t6) {
#pragma unroll
      for (int v = 0; v < 8; ++v) {
        int P = (mg * 6 + t6) * 16 + cm + v;
        int pos = i0 * 24 + P;
        y[((size_t)b * 577 + 1 + pos) * 64 + nt * 16 + cn] =
            f2bf(qgelu_f(acc[t6][v]));
      }
    }
  } else {  // CLS token: ycls[o] = sum_d xd[b,0,d] * w[b,o,d,1,1]  (tap 4)
    const int o = tid & 63, g = tid >> 6;
    const float* x0 = xd + (size_t)b * 577 * 64;
    const __bf16* wc = wperm + (((size_t)b * 9 + 4) * 64 + o) * 64;
    float s = 0;
    for (int d = g * 16; d < g * 16 + 16; ++d) s += x0[d] * bf2f(wc[d]);
    red[tid] = s;
    __syncthreads();
    if (tid < 64) {
      float t = red[tid] + red[tid + 64] + red[tid + 128] + red[tid + 192];
      y[(size_t)b * 577 * 64 + tid] = f2bf(qgelu_f(t));
    }
  }
}

// ---------- K5: out = y @ up_w^T + up_b + x ----------
__global__ void __launch_bounds__(512) k_up(
    const __bf16* __restrict__ y, const float* __restrict__ uw,
    const float* __restrict__ ub, const float* __restrict__ x,
    float* __restrict__ out) {
  __shared__ __bf16 Ws[128][64];
  const int tid = threadIdx.x;
  const int r0 = blockIdx.x * 64, n0 = blockIdx.y * 128;
  for (int idx = tid; idx < 8192; idx += 512) {
    int n = idx >> 6, k = idx & 63;
    Ws[n][k] = f2bf(uw[(size_t)(n0 + n) * 64 + k]);
  }
  __syncthreads();
  const int w = tid >> 5, lane = tid & 31;
  const int rt = w & 3, np = w >> 2;
  const int cm = (lane >> 4) << 3, cn = lane & 15;
  f32x8 acc[2] = {};
#pragma unroll
  for (int kt = 0; kt < 2; ++kt) {
    bf16x16 af = frag_bf16(y + (size_t)(r0 + rt * 16) * 64 + kt * 32, 64);
#pragma unroll
    for (int j = 0; j < 2; ++j) {
      bf16x16 bf = frag_bf16(&Ws[np * 32 + j * 16][0] + kt * 32, 64);
      acc[j] = wmma_bf16(af, bf, acc[j]);
    }
  }
#pragma unroll
  for (int j = 0; j < 2; ++j) {
    int ng = n0 + np * 32 + j * 16 + cn;
    float bias = ub[ng];
#pragma unroll
    for (int v = 0; v < 8; ++v) {
      int row = r0 + rt * 16 + cm + v;
      out[(size_t)row * 768 + ng] = acc[j][v] + bias + x[(size_t)row * 768 + ng];
    }
  }
}

// ---------- launcher ----------
extern "C" void kernel_launch(void* const* d_in, const int* in_sizes, int n_in,
                              void* d_out, int out_size, void* d_ws, size_t ws_size,
                              hipStream_t stream) {
  (void)in_sizes; (void)n_in; (void)out_size; (void)ws_size;
  const float* x   = (const float*)d_in[0];
  const float* mw1 = (const float*)d_in[1];
  const float* mb1 = (const float*)d_in[2];
  const float* mw2 = (const float*)d_in[3];
  const float* mb2 = (const float*)d_in[4];
  const float* mtw = (const float*)d_in[5];
  const float* hw  = (const float*)d_in[6];
  const float* hb  = (const float*)d_in[7];
  const float* dw  = (const float*)d_in[8];
  const float* db  = (const float*)d_in[9];
  const float* uw  = (const float*)d_in[10];
  const float* ub  = (const float*)d_in[11];
  float* out = (float*)d_out;

  // workspace layout (≈28.4 MB total)
  const size_t NP = (size_t)64 * 577 * 64;          // 2,363,392
  float* pf   = (float*)d_ws;                       // f32 (B,L,H)
  float* xd   = pf + NP;                            // f32 (B,L,D)
  float* feat = xd + NP;                            // f32 (B,H)
  __bf16* wperm = (__bf16*)(feat + 4096);           // bf16 (B,9,D,D)
  __bf16* yact  = wperm + (size_t)64 * 9 * 64 * 64; // bf16 (B,L,D)

  k_fused_in<<<577, 512, 0, stream>>>(x, mw1, mb1, mw2, mb2, dw, db, pf, xd);
  k_mask_feat<<<64, 256, 0, stream>>>(pf, mtw, feat);
  k_hyper<<<288, 512, 0, stream>>>(feat, hw, hb, wperm);
  k_conv_cls<<<dim3(64, 4), 256, 0, stream>>>(xd, wperm, yact);
  k_up<<<dim3(577, 6), 512, 0, stream>>>(yact, uw, ub, x, out);
}
